// GraphEmbeddings_7670811590986
// MI455X (gfx1250) — matrix operand order repaired and checked by reading, not compiled, not run on hardware
//
#include <hip/hip_runtime.h>

// ---------------------------------------------------------------------------
// GraphEmbeddings forward for MI455X (gfx1250, wave32, WMMA 16x16x32 f16)
// N=100000 nodes, E=1200000 edges, G=1000 graphs, D=64, 3 GIN layers.
// Memory-bound: feature maps (25.6MB) live in the 192MB L2; edge scatter uses
// global f32 atomics; all 64x64 GEMMs use v_wmma_f32_16x16x32_f16 with
// transposed-f16 weights (contiguous B fragments -> b128 loads).
// ---------------------------------------------------------------------------

#define NN 100000
#define EE 1200000
#define GG 1000
#define BN_EPS 1e-5f

typedef __attribute__((ext_vector_type(16))) _Float16 v16h;
typedef __attribute__((ext_vector_type(8)))  float    v8f;

// ---- WMMA fragment loaders (contiguous-k layouts, wave32) --------------

// A (16x32 f16) from a 16x64 f32 row-major tile in LDS, K-half kt (0/1).
// Per lane the fragment is one contiguous 16-float run: k=16g+e.
__device__ __forceinline__ v16h loadA_lds(const float* t, int lane, int kt) {
  int m = lane & 15, g = (lane >> 4) & 1;
  const float* p0 = t + m * 64 + kt * 32 + 16 * g;
  const float* p1 = p0 + 8;
  v16h a;
#pragma unroll
  for (int e = 0; e < 8; ++e) a[e] = (_Float16)p0[e];
#pragma unroll
  for (int e = 0; e < 8; ++e) a[8 + e] = (_Float16)p1[e];
  return a;
}

// B (32x16 f16) from TRANSPOSED f16 weight Wt[n][k] (64x64): per lane the
// fragment is 16 contiguous halfs (32B, 32B-aligned) -> vector loads.
__device__ __forceinline__ v16h loadB_t(const _Float16* Wt, int lane, int kt, int ct) {
  int n  = ct * 16 + (lane & 15);
  int kb = kt * 32 + ((lane >> 4) & 1) * 16;
  return *(const v16h*)(Wt + n * 64 + kb);
}

// ---- tiny utility kernels --------------------------------------------------

__global__ void zero_kernel(float* p, int n) {
  int i = blockIdx.x * blockDim.x + threadIdx.x;
  if (i < n) p[i] = 0.f;
}

// convert 64x64 f32 row-major W[k][n] -> f16 transposed Wt[n][k]
__global__ void cvtT_f16_kernel(const float* __restrict__ src, _Float16* __restrict__ dst) {
  int i = blockIdx.x * blockDim.x + threadIdx.x;
  if (i >= 4096) return;
  int k = i >> 6, n = i & 63;
  dst[n * 64 + k] = (_Float16)src[k * 64 + n];
}

// block1 first linear: h1[i][c] = x(i) dot W1[:,c] + b1[c], x = (cx, cy, depot)
__global__ void lin1_kernel(const float* __restrict__ coords, const int* __restrict__ depots,
                            const float* __restrict__ W1, const float* __restrict__ b1,
                            float* __restrict__ out, int n) {
  int idx = blockIdx.x * blockDim.x + threadIdx.x;
  if (idx >= n * 64) return;
  int i = idx >> 6, c = idx & 63;
  float x0 = coords[2 * i], x1 = coords[2 * i + 1];
  float x2 = depots[i] ? 1.f : 0.f;
  out[idx] = x0 * W1[c] + x1 * W1[64 + c] + x2 * W1[128 + c] + b1[c];
}

// per-channel sum / sum-of-squares of a rows x 64 matrix (for training BN)
__global__ void colsum_kernel(const float* __restrict__ x, int rows,
                              float* __restrict__ sum, float* __restrict__ sq) {
  __shared__ float s1[64], s2[64];
  if (threadIdx.x < 64) { s1[threadIdx.x] = 0.f; s2[threadIdx.x] = 0.f; }
  __syncthreads();
  int c = threadIdx.x & 63;
  int rpb = blockDim.x >> 6;
  float a = 0.f, b = 0.f;
  for (int r = blockIdx.x * rpb + (threadIdx.x >> 6); r < rows; r += gridDim.x * rpb) {
    float v = x[r * 64 + c];
    a += v; b += v * v;
  }
  atomicAdd(&s1[c], a);
  atomicAdd(&s2[c], b);
  __syncthreads();
  if (threadIdx.x < 64) {
    atomicAdd(&sum[threadIdx.x], s1[threadIdx.x]);
    atomicAdd(&sq[threadIdx.x],  s2[threadIdx.x]);
  }
}

// scale = g*rsqrt(var+eps), shift = beta - mean*scale   (biased var, training BN)
__global__ void bnfin_kernel(const float* sum, const float* sq,
                             const float* g, const float* be, float rows,
                             float* scale, float* shift) {
  int c = threadIdx.x;
  if (c >= 64) return;
  float mean = sum[c] / rows;
  float var  = sq[c] / rows - mean * mean;
  float sc   = g[c] * rsqrtf(var + BN_EPS);
  scale[c] = sc;
  shift[c] = be[c] - mean * sc;
}

__global__ void bn_relu_kernel(const float* __restrict__ pre, const float* __restrict__ scale,
                               const float* __restrict__ shift, float* __restrict__ out, int n) {
  int i = blockIdx.x * blockDim.x + threadIdx.x;
  if (i >= n) return;
  int c = i & 63;
  float v = pre[i] * scale[c] + shift[c];
  out[i] = v > 0.f ? v : 0.f;
}

// z = relu(bn(pre)); h = z; verts += z
__global__ void bn_relu_acc_kernel(const float* __restrict__ pre, const float* __restrict__ scale,
                                   const float* __restrict__ shift, float* __restrict__ h,
                                   float* __restrict__ verts, int n) {
  int i = blockIdx.x * blockDim.x + threadIdx.x;
  if (i >= n) return;
  int c = i & 63;
  float v = pre[i] * scale[c] + shift[c];
  v = v > 0.f ? v : 0.f;
  h[i] = v;
  verts[i] += v;
}

__global__ void bn_affine_kernel(const float* __restrict__ pre, const float* __restrict__ scale,
                                 const float* __restrict__ shift, float* __restrict__ out, int n) {
  int i = blockIdx.x * blockDim.x + threadIdx.x;
  if (i >= n) return;
  int c = i & 63;
  out[i] = pre[i] * scale[c] + shift[c];
}

__global__ void deg_kernel(const int* __restrict__ dst, float* __restrict__ deg, int e) {
  int i = blockIdx.x * blockDim.x + threadIdx.x;
  if (i < e) atomicAdd(&deg[dst[i]], 1.f);
}

// agg[dst] += h[src]  (float4 per thread, 16 threads per edge)
__global__ void scatter_kernel(const int* __restrict__ src, const int* __restrict__ dst,
                               const float* __restrict__ h, float* __restrict__ agg, int e) {
  int t = blockIdx.x * blockDim.x + threadIdx.x;
  if (t >= e * 16) return;
  int ed = t >> 4, c4 = (t & 15) << 2;
  int s = src[ed], d = dst[ed];
  const float4 v = *(const float4*)&h[(size_t)s * 64 + c4];
  float* p = &agg[(size_t)d * 64 + c4];
  atomicAdd(p + 0, v.x);
  atomicAdd(p + 1, v.y);
  atomicAdd(p + 2, v.z);
  atomicAdd(p + 3, v.w);
}

// pooled[batch[i]] += verts[i]; cnt[batch[i]] += 1
__global__ void pool_kernel(const float* __restrict__ verts, const int* __restrict__ batch,
                            float* __restrict__ pooled, float* __restrict__ cnt, int n) {
  int t = blockIdx.x * blockDim.x + threadIdx.x;
  if (t >= n * 16) return;
  int i = t >> 4, c4 = (t & 15) << 2;
  int b = batch[i];
  const float4 v = *(const float4*)&verts[(size_t)i * 64 + c4];
  float* p = &pooled[(size_t)b * 64 + c4];
  atomicAdd(p + 0, v.x);
  atomicAdd(p + 1, v.y);
  atomicAdd(p + 2, v.z);
  atomicAdd(p + 3, v.w);
  if ((t & 15) == 0) atomicAdd(&cnt[b], 1.f);
}

__global__ void pooldiv_kernel(float* pooled, const float* cnt, int g) {
  int i = blockIdx.x * blockDim.x + threadIdx.x;
  if (i >= g * 64) return;
  float c = cnt[i >> 6];
  pooled[i] /= (c > 1.f ? c : 1.f);
}

// ---- WMMA GEMM kernels -----------------------------------------------------

// out[rows x 64] = in[rows x 64] @ W + bias.  FULL: rows%16==0 so every wave's
// 16-row chunk is fully valid (no per-lane guards).  COPY: also mirror to cp.
template <bool FULL, bool COPY>
__global__ void __launch_bounds__(256)
gemm1_kernel(const float* __restrict__ in, const _Float16* __restrict__ Wt,
             const float* __restrict__ bias, float* __restrict__ out,
             float* __restrict__ cp, int rows) {
  __shared__ float tile[8][16 * 64];
  int wave = threadIdx.x >> 5, lane = threadIdx.x & 31;
  int row0 = (blockIdx.x * 8 + wave) * 16;
  if (row0 >= rows) return;                 // wave-uniform; EXEC stays all-ones
  float* t = tile[wave];
#pragma unroll
  for (int i = 0; i < 8; ++i) {             // 8 float4 per lane, coalesced
    int i4 = i * 32 + lane;
    int r = i4 >> 4, c4 = (i4 & 15) << 2;
    float4 v = make_float4(0.f, 0.f, 0.f, 0.f);
    if (FULL || row0 + r < rows) v = *(const float4*)&in[(size_t)(row0 + r) * 64 + c4];
    *(float4*)&t[r * 64 + c4] = v;
  }
  v16h a0 = loadA_lds(t, lane, 0);
  v16h a1 = loadA_lds(t, lane, 1);
  int n = lane & 15, g = (lane >> 4) & 1;
#pragma unroll
  for (int ct = 0; ct < 4; ++ct) {
    float bv = bias[ct * 16 + n];
    v8f c;
#pragma unroll
    for (int r = 0; r < 8; ++r) c[r] = bv;
    c = __builtin_amdgcn_wmma_f32_16x16x32_f16(false, a0, false, loadB_t(Wt, lane, 0, ct),
                                               (short)0, c, false, false);
    c = __builtin_amdgcn_wmma_f32_16x16x32_f16(false, a1, false, loadB_t(Wt, lane, 1, ct),
                                               (short)0, c, false, false);
    size_t o0 = (size_t)(row0 + 8 * g) * 64 + ct * 16 + n;
#pragma unroll
    for (int r = 0; r < 8; ++r) {
      if (FULL || row0 + 8 * g + r < rows) {
        out[o0 + (size_t)r * 64] = c[r];
        if (COPY) cp[o0 + (size_t)r * 64] = c[r];
      }
    }
  }
}

// zpre = relu((h + agg/max(deg,1)) @ W1 + b1) @ W2 + b2   (fused GIN MLP)
// rows % 16 == 0 (N-path only): no per-lane guards anywhere.
__global__ void __launch_bounds__(256)
mlp2_kernel(const float* __restrict__ h, const float* __restrict__ agg,
            const float* __restrict__ deg,
            const _Float16* __restrict__ W1t, const float* __restrict__ b1,
            const _Float16* __restrict__ W2t, const float* __restrict__ b2,
            float* __restrict__ out, int rows) {
  __shared__ float tile[8][16 * 64];
  int wave = threadIdx.x >> 5, lane = threadIdx.x & 31;
  int row0 = (blockIdx.x * 8 + wave) * 16;
  if (row0 >= rows) return;
  float* t = tile[wave];
#pragma unroll
  for (int i = 0; i < 8; ++i) {
    int i4 = i * 32 + lane;
    int r = i4 >> 4, c4 = (i4 & 15) << 2;
    size_t o = (size_t)(row0 + r) * 64 + c4;
    float d = deg[row0 + r];
    d = d > 1.f ? d : 1.f;
    float4 hv = *(const float4*)&h[o];
    float4 av = *(const float4*)&agg[o];
    float4 v;
    v.x = hv.x + av.x / d;
    v.y = hv.y + av.y / d;
    v.z = hv.z + av.z / d;
    v.w = hv.w + av.w / d;
    *(float4*)&t[r * 64 + c4] = v;
  }
  v16h a0 = loadA_lds(t, lane, 0);
  v16h a1 = loadA_lds(t, lane, 1);
  int n = lane & 15, g = (lane >> 4) & 1;

  v8f acc[4];
#pragma unroll
  for (int ct = 0; ct < 4; ++ct) {
    float bv = b1[ct * 16 + n];
    v8f c;
#pragma unroll
    for (int r = 0; r < 8; ++r) c[r] = bv;
    c = __builtin_amdgcn_wmma_f32_16x16x32_f16(false, a0, false, loadB_t(W1t, lane, 0, ct),
                                               (short)0, c, false, false);
    c = __builtin_amdgcn_wmma_f32_16x16x32_f16(false, a1, false, loadB_t(W1t, lane, 1, ct),
                                               (short)0, c, false, false);
    acc[ct] = c;
  }
  // relu, stage back to the LDS tile for the second GEMM's A-layout reload
#pragma unroll
  for (int ct = 0; ct < 4; ++ct)
#pragma unroll
    for (int r = 0; r < 8; ++r) {
      float v = acc[ct][r];
      t[(r + 8 * g) * 64 + ct * 16 + n] = v > 0.f ? v : 0.f;
    }
  a0 = loadA_lds(t, lane, 0);   // DS ops are in-order per wave (RAW safe)
  a1 = loadA_lds(t, lane, 1);
#pragma unroll
  for (int ct = 0; ct < 4; ++ct) {
    float bv = b2[ct * 16 + n];
    v8f c;
#pragma unroll
    for (int r = 0; r < 8; ++r) c[r] = bv;
    c = __builtin_amdgcn_wmma_f32_16x16x32_f16(false, a0, false, loadB_t(W2t, lane, 0, ct),
                                               (short)0, c, false, false);
    c = __builtin_amdgcn_wmma_f32_16x16x32_f16(false, a1, false, loadB_t(W2t, lane, 1, ct),
                                               (short)0, c, false, false);
    size_t o0 = (size_t)(row0 + 8 * g) * 64 + ct * 16 + n;
#pragma unroll
    for (int r = 0; r < 8; ++r) out[o0 + (size_t)r * 64] = c[r];
  }
}

// ---------------------------------------------------------------------------

static inline void zero_f(float* p, int n, hipStream_t s) {
  zero_kernel<<<(n + 255) / 256, 256, 0, s>>>(p, n);
}

extern "C" void kernel_launch(void* const* d_in, const int* in_sizes, int n_in,
                              void* d_out, int out_size, void* d_ws, size_t ws_size,
                              hipStream_t stream) {
  (void)in_sizes; (void)n_in; (void)out_size; (void)ws_size;
  const int N = NN, E = EE, G = GG;
  const size_t N64 = (size_t)N * 64;

  const float* coords = (const float*)d_in[0];
  const int*   eidx   = (const int*)d_in[1];   // [0..E): src, [E..2E): dst
  const int*   depots = (const int*)d_in[2];
  const int*   batch  = (const int*)d_in[3];
  const float* b1_W1 = (const float*)d_in[4];
  const float* b1_b1 = (const float*)d_in[5];
  const float* b1_g  = (const float*)d_in[6];
  const float* b1_be = (const float*)d_in[7];
  const float* b1_W2 = (const float*)d_in[8];
  const float* b1_b2 = (const float*)d_in[9];
  const float* msgW1[3], *msgb1[3], *msgW2[3], *msgb2[3], *msgg[3], *msgbe[3];
  for (int l = 0; l < 3; ++l) {
    msgW1[l] = (const float*)d_in[10 + 6 * l];
    msgb1[l] = (const float*)d_in[11 + 6 * l];
    msgW2[l] = (const float*)d_in[12 + 6 * l];
    msgb2[l] = (const float*)d_in[13 + 6 * l];
    msgg[l]  = (const float*)d_in[14 + 6 * l];
    msgbe[l] = (const float*)d_in[15 + 6 * l];
  }
  const float* b2_W  = (const float*)d_in[28];
  const float* b2_b  = (const float*)d_in[29];
  const float* b2_g  = (const float*)d_in[30];
  const float* b2_be = (const float*)d_in[31];

  float* verts = (float*)d_out;            // N x 64
  float* ctx   = verts + N64;              // G x 64

  // workspace layout
  float* h      = (float*)d_ws;            // N x 64
  float* agg    = h + N64;                 // N x 64
  float* zpre   = agg + N64;               // N x 64 (also ctx pre-BN, G x 64)
  float* deg    = zpre + N64;              // N
  float* pooled = deg + N;                 // G x 64
  float* cnt    = pooled + (size_t)G * 64; // G
  float* bnsum  = cnt + G;                 // 64
  float* bnsq   = bnsum + 64;              // 64
  float* scale  = bnsq + 64;               // 64
  float* shift  = scale + 64;              // 64
  // f16 transposed weights, aligned to 64B for 32B vector loads
  _Float16* wh = (_Float16*)(((uintptr_t)(shift + 64) + 63) & ~(uintptr_t)63);

  // convert + transpose the 8 64x64 weight matrices to f16 once
  const float* wsrc[8] = { b1_W2, msgW1[0], msgW2[0], msgW1[1], msgW2[1],
                           msgW1[2], msgW2[2], b2_W };
  for (int i = 0; i < 8; ++i)
    cvtT_f16_kernel<<<16, 256, 0, stream>>>(wsrc[i], wh + (size_t)i * 4096);

  // degrees (reused every layer)
  zero_f(deg, N, stream);
  deg_kernel<<<(E + 255) / 256, 256, 0, stream>>>(eidx + E, deg, E);

  const int gemmBlocks = ((N / 16) + 7) / 8;      // 6250 waves -> 782 blocks

  // ---- block1: Linear(3,64) -> BN -> ReLU -> Linear(64,64) ----
  lin1_kernel<<<(N * 64 + 255) / 256, 256, 0, stream>>>(coords, depots, b1_W1, b1_b1, zpre, N);
  zero_f(bnsum, 128, stream);
  colsum_kernel<<<256, 256, 0, stream>>>(zpre, N, bnsum, bnsq);
  bnfin_kernel<<<1, 64, 0, stream>>>(bnsum, bnsq, b1_g, b1_be, (float)N, scale, shift);
  bn_relu_kernel<<<(int)((N64 + 255) / 256), 256, 0, stream>>>(zpre, scale, shift, zpre, N * 64);
  gemm1_kernel<true, true><<<gemmBlocks, 256, 0, stream>>>(zpre, wh, b1_b2, h, verts, N);

  // ---- 3 GIN message layers ----
  for (int l = 0; l < 3; ++l) {
    zero_f(agg, (int)N64, stream);
    scatter_kernel<<<(E * 16 + 255) / 256, 256, 0, stream>>>(eidx, eidx + E, h, agg, E);
    mlp2_kernel<<<gemmBlocks, 256, 0, stream>>>(h, agg, deg,
                                                wh + (size_t)(1 + 2 * l) * 4096, msgb1[l],
                                                wh + (size_t)(2 + 2 * l) * 4096, msgb2[l],
                                                zpre, N);
    zero_f(bnsum, 128, stream);
    colsum_kernel<<<256, 256, 0, stream>>>(zpre, N, bnsum, bnsq);
    bnfin_kernel<<<1, 64, 0, stream>>>(bnsum, bnsq, msgg[l], msgbe[l], (float)N, scale, shift);
    bn_relu_acc_kernel<<<(int)((N64 + 255) / 256), 256, 0, stream>>>(zpre, scale, shift,
                                                                     h, verts, N * 64);
  }

  // ---- global mean pool + block2: Linear -> BN ----
  zero_f(pooled, G * 64 + G, stream);
  pool_kernel<<<(N * 16 + 255) / 256, 256, 0, stream>>>(verts, batch, pooled, cnt, N);
  pooldiv_kernel<<<(G * 64 + 255) / 256, 256, 0, stream>>>(pooled, cnt, G);
  int ctxBlocks = (((G + 15) / 16) + 7) / 8;
  gemm1_kernel<false, false><<<ctxBlocks, 256, 0, stream>>>(pooled, wh + (size_t)7 * 4096,
                                                            b2_b, zpre, nullptr, G);
  zero_f(bnsum, 128, stream);
  colsum_kernel<<<16, 256, 0, stream>>>(zpre, G, bnsum, bnsq);
  bnfin_kernel<<<1, 64, 0, stream>>>(bnsum, bnsq, b2_g, b2_be, (float)G, scale, shift);
  bn_affine_kernel<<<(G * 64 + 255) / 256, 256, 0, stream>>>(zpre, scale, shift, ctx, G * 64);
}
